// GraphAttention_4423816315057
// MI455X (gfx1250) — matrix operand order repaired and checked by reading, not compiled, not run on hardware
//
#include <hip/hip_runtime.h>
#include <hip/hip_bf16.h>

typedef __attribute__((ext_vector_type(16))) __bf16 v16bf;
typedef __attribute__((ext_vector_type(8)))  float  v8f;

#define B_   4
#define N_   2048
#define IN_  1024
#define OUT_ 512
#define M_   (B_ * N_)   // 8192 rows total

// -------------------------------------------------------------------------
// Kernel 0: f32 -> bf16 bulk conversion, 8 elements/thread.
// Native __bf16 casts let the backend use packed v_cvt; single 16B store.
// -------------------------------------------------------------------------
__global__ __launch_bounds__(256) void cvt_bf16_kernel(
    const float* __restrict__ src, unsigned short* __restrict__ dst, int n8)
{
    int i = blockIdx.x * blockDim.x + threadIdx.x;
    if (i >= n8) return;
    const float4* s = (const float4*)src + (size_t)i * 2;
    float4 a = s[0];
    float4 b = s[1];
    union { __bf16 h[8]; uint4 q; } o;
    o.h[0] = (__bf16)a.x; o.h[1] = (__bf16)a.y;
    o.h[2] = (__bf16)a.z; o.h[3] = (__bf16)a.w;
    o.h[4] = (__bf16)b.x; o.h[5] = (__bf16)b.y;
    o.h[6] = (__bf16)b.z; o.h[7] = (__bf16)b.w;
    ((uint4*)dst)[i] = o.q;
}

// -------------------------------------------------------------------------
// Kernel 1: h[m, d] = sum_k x[m,k] * fc_w[d,k] + fc_b[d]
// bf16 WMMA 16x16x32, fp32 accumulate. Wave tile = 32 rows x 64 cols
// (2 A-frags x 4 B-frags -> 8 WMMA per K-step, pure b128 loads, no VALU).
// Block = 256 threads = 8 waves -> block tile 256 x 64.
// Per-lane fragment layout per ISA 7.12.2:
//   A: row = lane&15, K = 8*(lane>>4)+{0..7} and 16+8*(lane>>4)+{0..7}
//   B: col = lane&15, K = 8*(lane>>4)+{0..7} and 16+8*(lane>>4)+{0..7} (same as A)
//   C: elem r -> row = r + 8*(lane>>4), col = lane&15
// -------------------------------------------------------------------------
union AF { v16bf v; uint4 q[2]; };

__global__ __launch_bounds__(256) void gemm_h_kernel(
    const unsigned short* __restrict__ xb, const unsigned short* __restrict__ wb,
    const float* __restrict__ bias, float* __restrict__ h)
{
    const int lane = threadIdx.x & 31;
    const int wave = threadIdx.x >> 5;
    const int kh   = lane >> 4;
    const int ln   = lane & 15;

    const int rowBase = blockIdx.x * 256 + wave * 32;
    const int colBase = blockIdx.y * 64;

    const unsigned short* xr0 = xb + (size_t)(rowBase + ln)      * IN_;
    const unsigned short* xr1 = xb + (size_t)(rowBase + 16 + ln) * IN_;
    const unsigned short* wr  = wb + (size_t)(colBase + ln)      * IN_;

    v8f acc[2][4] = {};

    for (int k = 0; k < IN_; k += 32) {
        AF a0, a1;
        a0.q[0] = *(const uint4*)(xr0 + k + 8 * kh);
        a0.q[1] = *(const uint4*)(xr0 + k + 16 + 8 * kh);
        a1.q[0] = *(const uint4*)(xr1 + k + 8 * kh);
        a1.q[1] = *(const uint4*)(xr1 + k + 16 + 8 * kh);

        #pragma unroll
        for (int t = 0; t < 4; ++t) {
            AF bf;
            const unsigned short* pb = wr + (size_t)t * 16 * IN_ + k;
            bf.q[0] = *(const uint4*)(pb + 8 * kh);
            bf.q[1] = *(const uint4*)(pb + 16 + 8 * kh);

            acc[0][t] = __builtin_amdgcn_wmma_f32_16x16x32_bf16(
                false, a0.v, false, bf.v, (short)0, acc[0][t], false, false);
            acc[1][t] = __builtin_amdgcn_wmma_f32_16x16x32_bf16(
                false, a1.v, false, bf.v, (short)0, acc[1][t], false, false);
        }
    }

    #pragma unroll
    for (int mi = 0; mi < 2; ++mi) {
        #pragma unroll
        for (int t = 0; t < 4; ++t) {
            const int   col = colBase + t * 16 + ln;
            const float bv  = bias[col];
            #pragma unroll
            for (int r = 0; r < 8; ++r) {
                const int row = rowBase + mi * 16 + r + 8 * kh;
                h[(size_t)row * OUT_ + col] = acc[mi][t][r] + bv;
            }
        }
    }
}

// -------------------------------------------------------------------------
// Kernel 2: s2[m] = dot(h[m, :], attn_w[OUT_ : 2*OUT_])   (one wave per row)
// (s1 and attn_b cancel under softmax shift-invariance -> not needed)
// -------------------------------------------------------------------------
__global__ __launch_bounds__(256) void s2_kernel(
    const float* __restrict__ h, const float* __restrict__ attn_w,
    float* __restrict__ s2)
{
    const int row  = (int)((blockIdx.x * blockDim.x + threadIdx.x) >> 5);
    const int lane = threadIdx.x & 31;
    if (row >= M_) return;

    const float* hr = h + (size_t)row * OUT_;
    const float* a2 = attn_w + OUT_;
    float acc = 0.0f;
    #pragma unroll 4
    for (int d = lane; d < OUT_; d += 32) acc += hr[d] * a2[d];
    #pragma unroll
    for (int off = 16; off > 0; off >>= 1) acc += __shfl_down(acc, off, 32);
    if (lane == 0) s2[row] = acc;
}

// -------------------------------------------------------------------------
// Kernel 3: per batch b: p = softmax(s2[b,:]);
//           row[d] = sum_j p[j] * h[b,j,d];  out[b,i,:] = row for all i.
// Grid = (4 batches, 2 column-chunks of 256), block = 256.
// -------------------------------------------------------------------------
__global__ __launch_bounds__(256) void attn_out_kernel(
    const float* __restrict__ h, const float* __restrict__ s2,
    float* __restrict__ out)
{
    __shared__ float p[N_];      // 8 KB
    __shared__ float red[256];

    const int b   = blockIdx.x;
    const int d0  = blockIdx.y * 256;
    const int tid = threadIdx.x;

    float lmax = -3.0e38f;
    for (int j = tid; j < N_; j += 256) {
        float v = s2[b * N_ + j];
        p[j] = v;
        lmax = fmaxf(lmax, v);
    }
    red[tid] = lmax;
    __syncthreads();
    for (int s = 128; s > 0; s >>= 1) {
        if (tid < s) red[tid] = fmaxf(red[tid], red[tid + s]);
        __syncthreads();
    }
    const float mx = red[0];
    __syncthreads();

    float lsum = 0.0f;
    for (int j = tid; j < N_; j += 256) {
        float e = __expf(p[j] - mx);
        p[j] = e;
        lsum += e;
    }
    __syncthreads();
    red[tid] = lsum;
    __syncthreads();
    for (int s = 128; s > 0; s >>= 1) {
        if (tid < s) red[tid] += red[tid + s];
        __syncthreads();
    }
    const float inv = 1.0f / red[0];
    __syncthreads();

    const int d = d0 + tid;
    const float* hb = h + (size_t)b * N_ * OUT_ + d;
    float acc = 0.0f;
    #pragma unroll 8
    for (int j = 0; j < N_; ++j) acc += p[j] * hb[(size_t)j * OUT_];
    acc *= inv;

    float* ob = out + (size_t)b * N_ * OUT_ + d;
    #pragma unroll 8
    for (int i = 0; i < N_; ++i) ob[(size_t)i * OUT_] = acc;
}

// -------------------------------------------------------------------------
extern "C" void kernel_launch(void* const* d_in, const int* in_sizes, int n_in,
                              void* d_out, int out_size, void* d_ws, size_t ws_size,
                              hipStream_t stream)
{
    const float* x      = (const float*)d_in[0];  // (4, 2048, 1024)
    const float* fc_w   = (const float*)d_in[1];  // (512, 1024)
    const float* fc_b   = (const float*)d_in[2];  // (512,)
    const float* attn_w = (const float*)d_in[3];  // (1, 1024)
    // d_in[4] = attn_b : cancels exactly under softmax shift-invariance

    float* out = (float*)d_out;

    // workspace layout
    float*          h    = (float*)d_ws;                       // 16 MB
    unsigned short* x_bf = (unsigned short*)(h + (size_t)M_ * OUT_);   // 16 MB
    unsigned short* w_bf = x_bf + (size_t)M_ * IN_;            // 1 MB
    float*          s2   = (float*)(w_bf + (size_t)OUT_ * IN_);// 32 KB

    // 0) bulk f32 -> bf16
    const int nx8 = (M_ * IN_) / 8;       // 1,048,576
    const int nw8 = (OUT_ * IN_) / 8;     // 65,536
    cvt_bf16_kernel<<<nx8 / 256, 256, 0, stream>>>(x, x_bf, nx8);
    cvt_bf16_kernel<<<nw8 / 256, 256, 0, stream>>>(fc_w, w_bf, nw8);

    // 1) WMMA GEMM
    dim3 g1(M_ / 256, OUT_ / 64);
    gemm_h_kernel<<<g1, 256, 0, stream>>>(x_bf, w_bf, fc_b, h);

    // 2) attention scores (only s2 matters)
    s2_kernel<<<M_ / 8, 256, 0, stream>>>(h, attn_w, s2);

    // 3) softmax + rank-1 weighted sum + broadcast
    dim3 g3(B_, OUT_ / 256);
    attn_out_kernel<<<g3, 256, 0, stream>>>(h, s2, out);
}
